// SplittedLora_59459527246488
// MI455X (gfx1250) — matrix operand order repaired and checked by reading, not compiled, not run on hardware
//
#include <hip/hip_runtime.h>

typedef _Float16 v8h  __attribute__((ext_vector_type(8)));
typedef _Float16 v16h __attribute__((ext_vector_type(16)));
typedef float    v8f  __attribute__((ext_vector_type(8)));

#define CB1   2048
#define CB2   2048
#define CMULT 4
#define CIN   4096
#define COUT  4096
#define CRNK  32
#define CNA1  32
#define NJOBS 4096      // CB1 + CB2 collapsed jobs (out-row == x-row == job id)
#define NGRP  64        // 32 seg-1 wid groups + 32 seg-2 j groups
#define MAXTILES 320    // <= NJOBS/16 + NGRP
#define KC    128       // K-chunk staged in LDS per iteration

// ws int layout
#define WS_COUNTS  0
#define WS_CURSOR  64
#define WS_OFFSETS 128
#define WS_NTILES  192
#define WS_TILES   256      // 3 ints per tile, up to 320 tiles
#define WS_ORDER   2048     // 4096 ints

__device__ __forceinline__ int job_group(int l, const int* __restrict__ wids) {
  if (l < CB1) return wids[l];                         // seg-1: adapter id 0..31
  int b = l - CB1;
  return CNA1 + ((wids[CB1 + b * CMULT] - CNA1) >> 2); // seg-2: j 0..31 -> group 32..63
}

__global__ void k_init(int* __restrict__ W) {
  int t = threadIdx.x;
  if (t < NGRP) { W[WS_COUNTS + t] = 0; W[WS_CURSOR + t] = 0; }
}

__global__ void k_count(const int* __restrict__ wids, int* __restrict__ W) {
  int l = blockIdx.x * 256 + threadIdx.x;
  if (l < NJOBS) atomicAdd(&W[WS_COUNTS + job_group(l, wids)], 1);
}

__global__ void k_build(int* __restrict__ W) {
  if (threadIdx.x != 0 || blockIdx.x != 0) return;
  int off = 0, nt = 0;
  for (int g = 0; g < NGRP; ++g) {
    int c = W[WS_COUNTS + g];
    W[WS_OFFSETS + g] = off;
    for (int s = 0; s < c; s += 16) {
      int rem = c - s;
      W[WS_TILES + nt * 3 + 0] = g;
      W[WS_TILES + nt * 3 + 1] = off + s;
      W[WS_TILES + nt * 3 + 2] = rem < 16 ? rem : 16;
      ++nt;
    }
    off += c;
  }
  W[WS_NTILES] = nt;
}

__global__ void k_scatter(const int* __restrict__ wids, int* __restrict__ W) {
  int l = blockIdx.x * 256 + threadIdx.x;
  if (l < NJOBS) {
    int g = job_group(l, wids);
    int pos = W[WS_OFFSETS + g] + atomicAdd(&W[WS_CURSOR + g], 1);
    W[WS_ORDER + pos] = l;   // order within group arbitrary; per-row result invariant
  }
}

#define SHUF16(lo, hi) __builtin_shufflevector(lo, hi, 0,1,2,3,4,5,6,7,8,9,10,11,12,13,14,15)

struct __align__(16) SM {
  union {
    _Float16 At[CMULT][CRNK][136];  // A^T chunk: [m][r][k], 272B rows (16B aligned, bank-spread)
    _Float16 Bt[CMULT][128][40];    // B^T chunk: [m][n][k], 80B rows
  } u;
  _Float16 Xc[16][KC];              // gathered X chunk (row-major, no transpose)
  _Float16 y16[CMULT][16][CRNK];    // per-tile y, f16
  float    yPart[8][16][16];        // per-wave partial y accumulators
  int      jr[16];                  // job row ids (x-row == out-row)
};

// one 16x16x32 step of phase A: acc += Xc(16x32 @ k=s*32) @ At[m] column tile rh
__device__ __forceinline__ v8f astep(const SM* p, int m, int rh, int s, int Ni, int hi,
                                     v8f acc) {
  // A-operand from LDS: lane holds K-chunks {0-7,16-23} (lo) / {8-15,24-31} (hi)
  const _Float16* ap = &p->Xc[Ni][s * 32 + hi * 8];
  v8h alo = *(const v8h*)ap;
  v8h ahi = *(const v8h*)(ap + 16);
  v16h a = SHUF16(alo, ahi);
  // B-operand: column r = rh*16+Ni of A_g; K {0-7,16-23} (lo) / {8-15,24-31} (hi)
  const _Float16* bp = &p->u.At[m][rh * 16 + Ni][s * 32 + hi * 8];
  v8h blo = *(const v8h*)bp;
  v8h bhi = *(const v8h*)(bp + 16);
  v16h b = SHUF16(blo, bhi);
  return __builtin_amdgcn_wmma_f32_16x16x32_f16(false, a, false, b, (short)0, acc,
                                                false, false);
}

// one 16x16x32 step of phase B: acc += y[m](16x32) @ Bt[m] tile
__device__ __forceinline__ v8f bstep(const SM* p, int m, int ncol, int Ni, int hi, v8f acc) {
  const _Float16* ap = &p->y16[m][Ni][hi * 8];
  v8h alo = *(const v8h*)ap;
  v8h ahi = *(const v8h*)(ap + 16);
  v16h a = SHUF16(alo, ahi);
  const _Float16* bp = &p->u.Bt[m][ncol + Ni][hi * 8];
  v8h blo = *(const v8h*)bp;
  v8h bhi = *(const v8h*)(bp + 16);
  v16h b = SHUF16(blo, bhi);
  return __builtin_amdgcn_wmma_f32_16x16x32_f16(false, a, false, b, (short)0, acc,
                                                false, false);
}

__launch_bounds__(256, 1)
__global__ void lora_main(const _Float16* __restrict__ x,
                          const _Float16* __restrict__ Aw,
                          const _Float16* __restrict__ Bw,
                          const int* __restrict__ W,
                          _Float16* __restrict__ out) {
  __shared__ SM sm;

  const int tb = blockIdx.x;
  if (tb >= W[WS_NTILES]) return;
  // force tile metadata into SGPRs
  const int g     = __builtin_amdgcn_readfirstlane(W[WS_TILES + tb * 3 + 0]);
  const int start = __builtin_amdgcn_readfirstlane(W[WS_TILES + tb * 3 + 1]);
  const int cnt   = __builtin_amdgcn_readfirstlane(W[WS_TILES + tb * 3 + 2]);
  const int nm    = (g < CNA1) ? 1 : CMULT;
  const int base  = (g < CNA1) ? g : CNA1 + (g - CNA1) * CMULT;
  const int* __restrict__ order = W + WS_ORDER;

  const int tid  = threadIdx.x;
  const int lane = tid & 31;
  const int wave = __builtin_amdgcn_readfirstlane(tid >> 5);  // wave-uniform scalar
  const int Ni   = lane & 15;
  const int hi   = (lane >> 4) & 1;

  // Per-lane output rows for D-matrix positions M = v + 8*hi
  int rowv[8];
#pragma unroll
  for (int v = 0; v < 8; ++v) {
    int M = v + hi * 8;
    rowv[v] = (M < cnt) ? order[start + M] : -1;
  }
  // job row ids (clamped for padded tile rows) -> LDS for cooperative X staging
  if (tid < 16) sm.jr[tid] = order[start + (tid < cnt ? tid : 0)];

  // ---------------- Phase A: y[m] = X(16xK) @ A_{base+m}(KxR), f32 acc ----------------
  const int ks  = (nm == 1) ? 4 : 1;         // K-split so all 8 waves have work
  const int m_w = wave / (2 * ks);           // all scalar (wave is readfirstlane'd)
  const int rh  = (wave / ks) & 1;
  const int kq  = wave & (ks - 1);

  v8f accY = {0.f, 0.f, 0.f, 0.f, 0.f, 0.f, 0.f, 0.f};
  for (int k0 = 0; k0 < CIN; k0 += KC) {
    __syncthreads();
    // Stage gathered X chunk (plain b128 copy): 256 vectors, one per thread
    {
      int row = tid >> 4, vq = tid & 15;
      v8h vv = *(const v8h*)(x + (size_t)sm.jr[row] * CIN + k0 + vq * 8);
      *(v8h*)&sm.Xc[row][vq * 8] = vv;
    }
    // Stage A^T chunk: read A row-major (k,r) coalesced, scatter transposed to LDS
    const int nVec = nm * KC * 4;            // v8h vectors (8 halves each)
    for (int i = tid; i < nVec; i += 256) {
      int m  = i / (KC * 4);
      int e  = i % (KC * 4);
      int k  = e >> 2;
      int rq = e & 3;
      v8h vv = *(const v8h*)(Aw + ((size_t)(base + m) * CIN + (k0 + k)) * CRNK + rq * 8);
#pragma unroll
      for (int j = 0; j < 8; ++j) sm.u.At[m][rq * 8 + j][k] = vv[j];
    }
    __syncthreads();
    // uniform branch on K-split; fully static step chains so loads pipeline ahead of WMMAs
    if (ks == 1) {
#pragma unroll
      for (int s = 0; s < KC / 32; ++s) accY = astep(&sm, m_w, rh, s, Ni, hi, accY);
    } else {
      accY = astep(&sm, m_w, rh, kq, Ni, hi, accY);
    }
  }
  __syncthreads();
#pragma unroll
  for (int v = 0; v < 8; ++v) sm.yPart[wave][v + hi * 8][Ni] = accY[v];
  __syncthreads();
  // Reduce K-split partials -> f16 y
  for (int i = tid; i < nm * 16 * CRNK; i += 256) {
    int m = i / (16 * CRNK);
    int e = i % (16 * CRNK);
    int M = e / CRNK;
    int r = e % CRNK;
    int rhh = r >> 4, rl = r & 15;
    float s = 0.f;
    for (int q = 0; q < ks; ++q) s += sm.yPart[m * 2 * ks + rhh * ks + q][M][rl];
    sm.y16[m][M][r] = (_Float16)s;
  }

  // ---------------- Phase B: out rows = 2 * sum_m y[m](16x32) @ B_{base+m}(32xN) ------
  for (int n0 = 0; n0 < COUT; n0 += 128) {
    __syncthreads();
    // Stage B^T chunk: read B row-major (k, n) coalesced, scatter transposed to LDS
    const int nVec = nm * CRNK * 16;
    for (int i = tid; i < nVec; i += 256) {
      int m  = i / (CRNK * 16);
      int e  = i % (CRNK * 16);
      int k  = e >> 4;
      int cq = e & 15;
      v8h vv = *(const v8h*)(Bw + ((size_t)(base + m) * CRNK + k) * COUT + n0 + cq * 8);
#pragma unroll
      for (int j = 0; j < 8; ++j) sm.u.Bt[m][cq * 8 + j][k] = vv[j];
    }
    __syncthreads();
    const int ncol = wave * 16;              // one 16-col tile per wave per chunk
    v8f acc = {0.f, 0.f, 0.f, 0.f, 0.f, 0.f, 0.f, 0.f};
    if (nm == 1) {                           // uniform branch: static WMMA chains
      acc = bstep(&sm, 0, ncol, Ni, hi, acc);
    } else {
#pragma unroll
      for (int m = 0; m < CMULT; ++m)        // MULT-sum via C-operand chaining
        acc = bstep(&sm, m, ncol, Ni, hi, acc);
    }
    const int col = n0 + ncol + Ni;
#pragma unroll
    for (int v = 0; v < 8; ++v) {
      if (rowv[v] >= 0)
        out[(size_t)rowv[v] * COUT + col] = (_Float16)(2.0f * acc[v]);
    }
  }
}

extern "C" void kernel_launch(void* const* d_in, const int* in_sizes, int n_in,
                              void* d_out, int out_size, void* d_ws, size_t ws_size,
                              hipStream_t stream) {
  (void)in_sizes; (void)n_in; (void)out_size; (void)ws_size;
  const _Float16* x  = (const _Float16*)d_in[0];
  const int*      wids = (const int*)d_in[2];   // xids implied by structure (row == job id)
  const _Float16* Aw = (const _Float16*)d_in[3];
  const _Float16* Bw = (const _Float16*)d_in[4];
  _Float16* out = (_Float16*)d_out;
  int* W = (int*)d_ws;

  k_init   <<<1, 256, 0, stream>>>(W);
  k_count  <<<NJOBS / 256, 256, 0, stream>>>(wids, W);
  k_build  <<<1, 1, 0, stream>>>(W);
  k_scatter<<<NJOBS / 256, 256, 0, stream>>>(wids, W);
  lora_main<<<MAXTILES, 256, 0, stream>>>(x, Aw, Bw, W, out);
}
